// EpistemicMultiHeadAttention_78005196030553
// MI455X (gfx1250) — matrix operand order repaired and checked by reading, not compiled, not run on hardware
//
#include <hip/hip_runtime.h>

// ---------------------------------------------------------------------------
// EpistemicMultiHeadAttention for MI455X (gfx1250, wave32, WMMA bf16 + TDM)
// B=4, T=1024, D=1024, H=16, DK=64
// ---------------------------------------------------------------------------

enum { NB = 4, NT = 1024, ND = 1024, NH = 16, NDK = 64 };

typedef __attribute__((ext_vector_type(8)))  float        v8f;
typedef __attribute__((ext_vector_type(8)))  unsigned int v8u;
typedef __attribute__((ext_vector_type(16))) __bf16       v16bf;
typedef __attribute__((ext_vector_type(4)))  unsigned int u32x4;
typedef __attribute__((ext_vector_type(4)))  int          i32x4;
typedef __attribute__((ext_vector_type(8)))  int          i32x8;

#if defined(__has_builtin)
#  if __has_builtin(__builtin_amdgcn_tensor_load_to_lds)
#    define HAVE_TDM 1
#  endif
#endif
#ifndef HAVE_TDM
#  define HAVE_TDM 0
#endif

static __device__ __forceinline__ v8f wmma_bf16(const v8u& a, const v8u& b, const v8f& c) {
    return __builtin_amdgcn_wmma_f32_16x16x32_bf16(
        false, __builtin_bit_cast(v16bf, a),
        false, __builtin_bit_cast(v16bf, b),
        (short)0, c, false, false);
}

static __device__ __forceinline__ unsigned short f2bf(float f) {
    unsigned int u = __float_as_uint(f);
    u += 0x7fffu + ((u >> 16) & 1u);          // round to nearest even
    return (unsigned short)(u >> 16);
}

// A-fragment (16x32 bf16): lane half hf, VGPR v -> K base offset
static __device__ __forceinline__ int a_kbase(int v, int hf) {
    return hf * 16 + 2 * v;
}
// B-fragment (32x16 bf16): lane half hf, VGPR v -> K base offset
static __device__ __forceinline__ int b_kbase(int v, int hf) {
    return hf * 16 + 2 * v;
}

// ---------------------------------------------------------------------------
// TDM: issue a tensor_load_to_lds of a 16(row) x 32(col) bf16 tile.
//   global tile start = src (byte address), row stride = 1024 elements
//   LDS destination   = lds_byte (tile stored contiguously, 64B per row)
// D# layout per CDNA5 ISA (groups 0/1; groups 2/3 unused).
// ---------------------------------------------------------------------------
#if HAVE_TDM
static __device__ __forceinline__ void tdm_load_tile16x32(const unsigned short* src,
                                                          unsigned lds_byte) {
    unsigned long long ga = (unsigned long long)(size_t)(const void*)src;
    u32x4 g0;
    g0[0] = 1u;                                            // count=1, user descriptor
    g0[1] = lds_byte;                                      // lds_addr
    g0[2] = (unsigned)(ga & 0xffffffffull);                // global_addr[31:0]
    g0[3] = (unsigned)((ga >> 32) & 0x01ffffffull) | 0x80000000u; // addr[56:32] | type=2

    i32x8 g1;
    g1[0] = 0x00010000;            // wg_mask=0, data_size=1 (2B), no iterate/pad
    g1[1] = (int)(1024u << 16);    // tensor_dim0[15:0]=1024 (bits 63:48)
    g1[2] = (int)0xFFFF0000u;      // tensor_dim0 hi=0; tensor_dim1[15:0]=0xFFFF
    g1[3] = (int)(32u << 16);      // tensor_dim1 hi=0; tile_dim0=32 (bits 127:112)
    g1[4] = 16;                    // tile_dim1=16, tile_dim2=0
    g1[5] = 1024;                  // tensor_dim0_stride[31:0] = 1024 elements
    g1[6] = 0;                     // stride hi, tensor_dim1_stride lo
    g1[7] = 0;

    i32x4 gz = {0, 0, 0, 0};
#if __clang_major__ >= 23
    i32x8 gz8 = {0, 0, 0, 0, 0, 0, 0, 0};
    __builtin_amdgcn_tensor_load_to_lds(g0, g1, gz, gz, gz8, 0);
#else
    __builtin_amdgcn_tensor_load_to_lds(g0, g1, gz, gz, 0);
#endif
}
#endif

// ---------------------------------------------------------------------------
// fp32 -> bf16 elementwise conversion
// ---------------------------------------------------------------------------
__global__ void cvt_bf16_kernel(const float* __restrict__ src,
                                unsigned short* __restrict__ dst, int n) {
    int i = blockIdx.x * blockDim.x + threadIdx.x;
    if (i < n) dst[i] = f2bf(src[i]);
}

// ---------------------------------------------------------------------------
// Confidence gates (full fp32 path)
// ---------------------------------------------------------------------------
__global__ void gates_kernel(const float* __restrict__ x,
                             const float* __restrict__ Wq,
                             const float* __restrict__ bq,
                             const float* __restrict__ Wg1, const float* __restrict__ bg1,
                             const float* __restrict__ Wg2, const float* __restrict__ bg2,
                             float* __restrict__ cbuf, float* __restrict__ u_out) {
    __shared__ float ctx[ND];
    __shared__ float red1[256];
    __shared__ float red2[256];
    const int tid = threadIdx.x;
    const int b   = blockIdx.x;
    const float* xr = x + (size_t)b * NT * ND;   // row t = 0
    for (int j = tid; j < ND; j += 256) {
        float s = bq[j];
        const float* wr = Wq + (size_t)j * ND;
        for (int k = 0; k < ND; ++k) s += xr[k] * wr[k];
        ctx[j] = s;
    }
    __syncthreads();
    float p1 = 0.f, p2 = 0.f;
    for (int j = tid; j < ND; j += 256) { p1 += ctx[j] * Wg1[j]; p2 += ctx[j] * Wg2[j]; }
    red1[tid] = p1; red2[tid] = p2;
    __syncthreads();
    for (int s = 128; s > 0; s >>= 1) {
        if (tid < s) { red1[tid] += red1[tid + s]; red2[tid] += red2[tid + s]; }
        __syncthreads();
    }
    if (tid == 0) {
        float g1 = 1.f / (1.f + __expf(-(red1[0] + bg1[0])));
        float g2 = 1.f / (1.f + __expf(-(red2[0] + bg2[0])));
        float c  = fminf(fmaxf(g1 * g2, 1e-8f), 1.f);
        cbuf[b]     = c;
        cbuf[4 + b] = (c < 0.3f) ? c : 1.f;   // 1/tau
        u_out[b]    = 1.f - c;
    }
}

// ---------------------------------------------------------------------------
// Tiled bf16 GEMM: C[M=4096, N=1024] = A[M,K=1024] * W[N,K]^T + bias
// Block = 128 threads (4 waves); block tile 64(M) x 64(N).
// Wave w owns M rows [tm*64+w*16, +16) and all 64 N columns (4 accumulators),
// so each A fragment feeds 4 WMMAs. W tiles (64x32 bf16) are staged in LDS
// via per-wave TDM tensor_load_to_lds, double-buffered across K.
// mode 0: bf16 out, head-major [B,H,T,DK], value = (acc+bias)*scale
// mode 1: fp32 out, row-major [M,N]
// ---------------------------------------------------------------------------
__global__ void gemm_bf16_kernel(const unsigned short* __restrict__ A,
                                 const unsigned short* __restrict__ W,
                                 const float* __restrict__ bias,
                                 float scale, int mode,
                                 unsigned short* __restrict__ dst_bf,
                                 float* __restrict__ dst_f) {
    __shared__ unsigned short wtile[2][64 * 32];   // 8 KB double buffer
    const int tid  = threadIdx.x;
    const int lane = tid & 31;
    const int wv   = tid >> 5;                     // wave 0..3
    const int hf   = lane >> 4, mn = lane & 15;
    const int tn   = blockIdx.x;                   // N/64
    const int tm   = blockIdx.y;                   // M/64
    const size_t arow = (size_t)(tm * 64 + wv * 16 + mn) * ND;
    const size_t wrow0 = (size_t)(tn * 64 + wv * 16) * ND;   // this wave's W chunk

#if !HAVE_TDM
    const unsigned int* Wu = (const unsigned int*)W;
#endif

    // stage W tile for kk into wtile[bi] (each wave moves its own 16x32 chunk)
    auto stage = [&](int bi, int kk) {
#if HAVE_TDM
        unsigned lds = (unsigned)(size_t)(void*)&wtile[bi][wv * 512];
        tdm_load_tile16x32(W + wrow0 + kk, lds);
#else
        unsigned int* dstu = (unsigned int*)&wtile[bi][wv * 512];
        for (int i = lane; i < 256; i += 32) {
            int r = i >> 4, c2 = i & 15;
            dstu[i] = Wu[(wrow0 + (size_t)r * ND + kk) / 2 + c2];
        }
#endif
    };
    auto stage_wait = [&]() {
#if HAVE_TDM
        __builtin_amdgcn_s_wait_tensorcnt(0);
#endif
        __syncthreads();
    };

    stage(0, 0);
    stage_wait();

    v8f acc[4] = {{}, {}, {}, {}};
    int bi = 0;
    for (int kk = 0; kk < ND; kk += 32) {
        if (kk + 32 < ND) stage(bi ^ 1, kk + 32);   // prefetch next W tile

        v8u a;
#pragma unroll
        for (int v = 0; v < 8; ++v)
            a[v] = *(const unsigned int*)(A + arow + kk + a_kbase(v, hf));

#pragma unroll
        for (int j = 0; j < 4; ++j) {
            const unsigned int* bu = (const unsigned int*)&wtile[bi][(j * 16 + mn) * 32];
            v8u b;
#pragma unroll
            for (int v = 0; v < 8; ++v)
                b[v] = bu[b_kbase(v, hf) >> 1];
            acc[j] = wmma_bf16(a, b, acc[j]);
        }

        stage_wait();      // next buffer ready + all reads of current done
        bi ^= 1;
    }

#pragma unroll
    for (int j = 0; j < 4; ++j) {
        const int   ncol = tn * 64 + j * 16 + mn;
        const float bs   = bias[ncol];
        if (mode == 0) {
            const int hh = ncol >> 6, dc = ncol & 63;
#pragma unroll
            for (int r = 0; r < 8; ++r) {
                int row = tm * 64 + wv * 16 + r + 8 * hf;
                int bb = row >> 10, tt = row & (NT - 1);
                dst_bf[(((size_t)bb * NH + hh) * NT + tt) * NDK + dc] =
                    f2bf((acc[j][r] + bs) * scale);
            }
        } else {
#pragma unroll
            for (int r = 0; r < 8; ++r) {
                int row = tm * 64 + wv * 16 + r + 8 * hf;
                dst_f[(size_t)row * ND + ncol] = acc[j][r] + bs;
            }
        }
    }
}

// ---------------------------------------------------------------------------
// Attention: one wave per (b, h, 16-row q tile).  (unchanged from round 1)
// ---------------------------------------------------------------------------
__global__ void attn_kernel(const unsigned short* __restrict__ Qb,
                            const unsigned short* __restrict__ Kb,
                            const unsigned short* __restrict__ Vb,
                            const float* __restrict__ cbuf,
                            unsigned short* __restrict__ attnb) {
    __shared__ float S[16 * 1024];                 // 64 KB
    unsigned short* Pb = (unsigned short*)S;       // in-place bf16 probs
    const int lane = threadIdx.x;
    const int hf = lane >> 4, mn = lane & 15;
    const int mt = blockIdx.x, h = blockIdx.y, b = blockIdx.z;
    const size_t bh = (size_t)b * NH + h;
    const float cgate = cbuf[b];
    const float itau  = cbuf[4 + b];

    v8u aq0, aq1;
    const size_t qrow = (bh * NT + mt * 16 + mn) * NDK;
#pragma unroll
    for (int v = 0; v < 8; ++v) {
        int kb = a_kbase(v, hf);
        aq0[v] = *(const unsigned int*)(Qb + qrow + kb);
        aq1[v] = *(const unsigned int*)(Qb + qrow + 32 + kb);
    }

    for (int nt2 = 0; nt2 < NT / 16; ++nt2) {
        const size_t krow = (bh * NT + nt2 * 16 + mn) * NDK;
        v8u b0, b1;
#pragma unroll
        for (int v = 0; v < 8; ++v) {
            int kb = b_kbase(v, hf);
            b0[v] = *(const unsigned int*)(Kb + krow + kb);
            b1[v] = *(const unsigned int*)(Kb + krow + 32 + kb);
        }
        v8f s = {};
        s = wmma_bf16(aq0, b0, s);
        s = wmma_bf16(aq1, b1, s);
#pragma unroll
        for (int r = 0; r < 8; ++r)
            S[(r + 8 * hf) * 1024 + nt2 * 16 + mn] = s[r] * itau;
    }

    if (lane < 16) {
        const int r = lane;
        float mx = -3.4e38f;
        for (int j = 0; j < NT; ++j) mx = fmaxf(mx, S[r * 1024 + j]);
        float sum = 0.f;
        for (int j = 0; j < NT; ++j) {
            float e = __expf(S[r * 1024 + j] - mx);
            S[r * 1024 + j] = e;
            sum += e;
        }
        const float inv  = cgate / sum;
        const float base = (1.f - cgate) * (1.f / (float)NT);
        for (int j = 0; j < NT; ++j)
            Pb[r * 2048 + j] = f2bf(S[r * 1024 + j] * inv + base);
    }

    v8f oacc[4] = {{}, {}, {}, {}};
    const unsigned int* Pu = (const unsigned int*)S;
    for (int kk2 = 0; kk2 < NT; kk2 += 32) {
        v8u ap;
#pragma unroll
        for (int v = 0; v < 8; ++v)
            ap[v] = Pu[mn * 1024 + ((kk2 + a_kbase(v, hf)) >> 1)];
#pragma unroll
        for (int j = 0; j < 4; ++j) {
            v8u bvv;
#pragma unroll
            for (int v = 0; v < 8; ++v) {
                size_t t0 = bh * NT + kk2 + b_kbase(v, hf);
                unsigned int lo = Vb[t0 * NDK + j * 16 + mn];
                unsigned int hi = Vb[(t0 + 1) * NDK + j * 16 + mn];
                bvv[v] = lo | (hi << 16);
            }
            oacc[j] = wmma_bf16(ap, bvv, oacc[j]);
        }
    }

#pragma unroll
    for (int j = 0; j < 4; ++j) {
#pragma unroll
        for (int r = 0; r < 8; ++r) {
            int row = mt * 16 + r + 8 * hf;
            int col = j * 16 + mn;
            attnb[((size_t)b * NT + row) * ND + h * NDK + col] = f2bf(oacc[j][r]);
        }
    }
}

// ---------------------------------------------------------------------------
// Host-side launcher
// ---------------------------------------------------------------------------
extern "C" void kernel_launch(void* const* d_in, const int* in_sizes, int n_in,
                              void* d_out, int out_size, void* d_ws, size_t ws_size,
                              hipStream_t stream) {
    (void)in_sizes; (void)n_in; (void)out_size; (void)ws_size;
    const float* x   = (const float*)d_in[0];
    const float* Wq  = (const float*)d_in[1];
    const float* bq  = (const float*)d_in[2];
    const float* Wk  = (const float*)d_in[3];
    const float* bk  = (const float*)d_in[4];
    const float* Wv  = (const float*)d_in[5];
    const float* bv  = (const float*)d_in[6];
    const float* Wo  = (const float*)d_in[7];
    const float* bo  = (const float*)d_in[8];
    const float* Wg1 = (const float*)d_in[9];
    const float* bg1 = (const float*)d_in[10];
    const float* Wg2 = (const float*)d_in[11];
    const float* bg2 = (const float*)d_in[12];

    const size_t SZX = (size_t)NB * NT * ND;   // 4,194,304
    const size_t SZW = (size_t)ND * ND;        // 1,048,576

    char* ws = (char*)d_ws;
    unsigned short* xb  = (unsigned short*)ws;
    unsigned short* wqb = (unsigned short*)(ws + SZX * 2);
    unsigned short* wkb = wqb + SZW;
    unsigned short* wvb = wkb + SZW;
    unsigned short* wob = wvb + SZW;
    unsigned short* qb  = wob + SZW;
    unsigned short* kb  = qb + SZX;
    unsigned short* vb  = kb + SZX;
    unsigned short* ab  = vb + SZX;
    float* cbuf = (float*)(ab + SZX);

    float* outp  = (float*)d_out;
    float* u_out = outp + SZX;

    cvt_bf16_kernel<<<(unsigned)((SZX + 255) / 256), 256, 0, stream>>>(x,  xb,  (int)SZX);
    cvt_bf16_kernel<<<(unsigned)((SZW + 255) / 256), 256, 0, stream>>>(Wq, wqb, (int)SZW);
    cvt_bf16_kernel<<<(unsigned)((SZW + 255) / 256), 256, 0, stream>>>(Wk, wkb, (int)SZW);
    cvt_bf16_kernel<<<(unsigned)((SZW + 255) / 256), 256, 0, stream>>>(Wv, wvb, (int)SZW);
    cvt_bf16_kernel<<<(unsigned)((SZW + 255) / 256), 256, 0, stream>>>(Wo, wob, (int)SZW);

    gates_kernel<<<NB, 256, 0, stream>>>(x, Wq, bq, Wg1, bg1, Wg2, bg2, cbuf, u_out);

    dim3 gg(ND / 64, (NB * NT) / 64);   // (16, 64) blocks of 128 threads
    gemm_bf16_kernel<<<gg, 128, 0, stream>>>(xb, wqb, bq, 0.125f, 0, qb, nullptr);
    gemm_bf16_kernel<<<gg, 128, 0, stream>>>(xb, wkb, bk, 1.0f,   0, kb, nullptr);
    gemm_bf16_kernel<<<gg, 128, 0, stream>>>(xb, wvb, bv, 1.0f,   0, vb, nullptr);

    attn_kernel<<<dim3(NT / 16, NH, NB), 32, 0, stream>>>(qb, kb, vb, cbuf, ab);

    gemm_bf16_kernel<<<gg, 128, 0, stream>>>(ab, wob, bo, 1.0f, 1, nullptr, outp);
}
